// SelfAttention_88064009437613
// MI455X (gfx1250) — compile-verified, matrix operand order checked by reading, not run on hardware
//
#include <hip/hip_runtime.h>
#include <hip/hip_bf16.h>

// ---------------------------------------------------------------------------
// MHA for MI455X (gfx1250): bf16 WMMA everywhere, fp32 accumulate.
// N=2, L=2048, EMBED=1024, HEADS=16, HEAD_DIM=64, scale = 1/sqrt(1024) = 1/32
// ---------------------------------------------------------------------------

typedef __attribute__((ext_vector_type(16))) __bf16 v16bf;
typedef __attribute__((ext_vector_type(8)))  __bf16 v8bf;
typedef __attribute__((ext_vector_type(8)))  float  v8f;

// async global->LDS copy operand type (from clang's builtin signature:
// "int __vector(4) addrspace(1)*" / addrspace(3) for the LDS side)
typedef int v4i_async __attribute__((vector_size(4 * sizeof(int))));
typedef __attribute__((address_space(1))) v4i_async* gvec_p;
typedef __attribute__((address_space(3))) v4i_async* lvec_p;

#define NB    2
#define LSEQ  2048
#define EMB   1024
#define NHEAD 16
#define HDIM  64
#define ROWS  (NB * LSEQ)     // 4096
#define SCALE 0.03125f        // 1/sqrt(1024)

#define BSTRIDE 40            // padded LDS row stride (bf16 elems) for B slab

#if defined(__has_builtin)
#if __has_builtin(__builtin_amdgcn_global_load_async_to_lds_b128)
#define HAVE_ASYNC_LDS 1
#endif
#if __has_builtin(__builtin_amdgcn_s_wait_asynccnt)
#define HAVE_WAIT_ASYNC 1
#endif
#endif

// Load a 16x32 bf16 operand tile slice for one lane: two contiguous 8-elem
// runs (ISA 16-bit A-matrix layout; p must already include the +8*half skew).
static __device__ __forceinline__ v16bf ld_tile(const __bf16* p) {
    v8bf lo = *(const v8bf*)(p);
    v8bf hi = *(const v8bf*)(p + 16);
    return __builtin_shufflevector(lo, hi, 0,1,2,3,4,5,6,7,8,9,10,11,12,13,14,15);
}

static __device__ __forceinline__ v8f wmma_bf16(v16bf a, v16bf b, v8f c) {
    // (neg_a, A, neg_b, B, c_mod, C, reuse_a, reuse_b)
    return __builtin_amdgcn_wmma_f32_16x16x32_bf16(false, a, false, b,
                                                   (short)0, c, false, false);
}

// 16-byte async global->LDS copy (ASYNCcnt-tracked); sync fallback otherwise.
static __device__ __forceinline__ void async_copy16(const __bf16* g, __bf16* l) {
#if defined(HAVE_ASYNC_LDS)
    __builtin_amdgcn_global_load_async_to_lds_b128(
        (gvec_p)(v4i_async*)g, (lvec_p)(v4i_async*)l, 0, 0);
#else
    *(v8bf*)l = *(const v8bf*)g;
#endif
}

static __device__ __forceinline__ void wait_async_copies() {
#if defined(HAVE_ASYNC_LDS)
#if defined(HAVE_WAIT_ASYNC)
    __builtin_amdgcn_s_wait_asynccnt(0);
#else
    asm volatile("s_wait_asynccnt 0x0" ::: "memory");
#endif
#endif
}

// ---------------------------------------------------------------------------
// fp32 -> bf16 conversion (vectorized x4)
// ---------------------------------------------------------------------------
__global__ void __launch_bounds__(256)
cvt_f32_bf16(const float* __restrict__ in, __bf16* __restrict__ out, int n) {
    int idx = (blockIdx.x * blockDim.x + threadIdx.x) * 4;
    if (idx < n) {
        float4 v = *(const float4*)(in + idx);
        out[idx + 0] = (__bf16)v.x;
        out[idx + 1] = (__bf16)v.y;
        out[idx + 2] = (__bf16)v.z;
        out[idx + 3] = (__bf16)v.w;
    }
}

// ---------------------------------------------------------------------------
// GEMM: out[M,Nn] = A[M,K] * B[Nn,K]^T + bias  (torch Linear convention)
// Block = 8 waves computing a 128x64 tile. The 64x32 weight slab for each
// k-step is staged into LDS with CDNA5 async global->LDS copies (ASYNCcnt),
// double buffered: wait(prev) -> barrier -> issue(next) -> 4 WMMAs from LDS.
// ---------------------------------------------------------------------------
__global__ void __launch_bounds__(256)
gemm_bf16(const __bf16* __restrict__ A, const __bf16* __restrict__ B,
          const float* __restrict__ bias,
          __bf16* __restrict__ outb, float* __restrict__ outf,
          int M, int Nn, int K) {
    __shared__ __bf16 bs[2][64 * BSTRIDE];   // 2 x 5KB weight slabs

    const int lane = threadIdx.x & 31;
    const int w    = threadIdx.x >> 5;       // wave in block: row tile
    const int half = lane >> 4;
    const int lidx = lane & 15;

    const int colStrips = Nn >> 6;
    const int rowBlock  = blockIdx.x / colStrips;   // 128-row block
    const int colBase   = (blockIdx.x % colStrips) << 6;

    // staging assignment: each thread owns one 16B chunk of the 64x32 slab
    const int scol = threadIdx.x >> 2;              // 0..63
    const int skof = (threadIdx.x & 3) * 8;         // 0,8,16,24
    const __bf16* gB = B + (size_t)(colBase + scol) * K + skof;
    __bf16* lB[2] = { &bs[0][scol * BSTRIDE + skof], &bs[1][scol * BSTRIDE + skof] };

    const int row = rowBlock * 128 + w * 16 + lidx;  // A-layout: lane = row M
    const __bf16* aRow = A + (size_t)row * K + 8 * half;

    v8f acc[4] = {};

    // prologue: stage slab for k=0 into buffer 0
    async_copy16(gB, lB[0]);

    const int nk = K >> 5;
    for (int kk = 0; kk < nk; ++kk) {
        const int k  = kk << 5;
        const int cb = kk & 1;
        wait_async_copies();
        __syncthreads();                      // slab cb ready for everyone

        if (kk + 1 < nk) {                    // stage next slab into other buf
            async_copy16(gB + k + 32, lB[cb ^ 1]);
        }

        v16bf a = ld_tile(aRow + k);
#pragma unroll
        for (int t = 0; t < 4; ++t) {         // B-layout: lane = col N
            v16bf b = ld_tile(&bs[cb][(t * 16 + lidx) * BSTRIDE + 8 * half]);
            acc[t] = wmma_bf16(a, b, acc[t]);
        }
    }

#pragma unroll
    for (int t = 0; t < 4; ++t) {
        const int col = colBase + t * 16 + lidx;
        const float bv = bias[col];
#pragma unroll
        for (int r = 0; r < 8; ++r) {
            const int orow = rowBlock * 128 + w * 16 + r + 8 * half;  // C/D layout
            const float v = acc[t][r] + bv;
            if (outf) outf[(size_t)orow * Nn + col] = v;
            else      outb[(size_t)orow * Nn + col] = (__bf16)v;
        }
    }
}

// ---------------------------------------------------------------------------
// Causal flash attention: one wave per (batch, head, 16-query tile).
// Streams 32 keys per iteration: S = Q*K^T (4 WMMAs), online softmax,
// P transposed through per-wave LDS, O += P*V (4 WMMAs).
// ---------------------------------------------------------------------------
__global__ void __launch_bounds__(256)
attn_fwd(const __bf16* __restrict__ Q, const __bf16* __restrict__ Kb,
         const __bf16* __restrict__ Vb, __bf16* __restrict__ ctx) {
    // per wave: pT[16][32] + vT[64][32] bf16 = 5120 B; 8 waves = 40 KB
    __shared__ __bf16 smem[8 * 2560];

    const int wid  = (blockIdx.x * blockDim.x + threadIdx.x) >> 5;
    const int lane = threadIdx.x & 31;
    const int half = lane >> 4;
    const int lidx = lane & 15;
    const int wInB = (threadIdx.x >> 5);

    const int qTile = wid & 127;          // L/16 = 128
    const int head  = (wid >> 7) & 15;
    const int batch = wid >> 11;
    if (batch >= NB) return;

    __bf16* pT = smem + wInB * 2560;      // [16][32]
    __bf16* vT = smem + wInB * 2560 + 512;// [64][32] (dim-major, transposed V)

    // Q tile (16x64) in registers, A-layout (head dim contiguous in memory)
    const __bf16* qp = Q + ((size_t)(batch * LSEQ + qTile * 16 + lidx)) * EMB
                         + head * HDIM + 8 * half;
    const v16bf qa0 = ld_tile(qp);
    const v16bf qa1 = ld_tile(qp + 32);

    float mrun[8], lrun[8];
#pragma unroll
    for (int r = 0; r < 8; ++r) { mrun[r] = -3.0e38f; lrun[r] = 0.0f; }
    v8f o[4] = {};

    const int nb = (qTile * 16 + 47) >> 5;   // causal: key blocks of 32
    for (int jb = 0; jb < nb; ++jb) {
        const int j = jb << 5;

        // --- S = Q * K^T : two 16x16 tiles over keys [j,j+16) and [j+16,j+32)
        const __bf16* kp0 = Kb + ((size_t)(batch * LSEQ + j + lidx)) * EMB
                               + head * HDIM + 8 * half;
        const __bf16* kp1 = kp0 + (size_t)16 * EMB;
        v8f s0 = wmma_bf16(qa0, ld_tile(kp0), (v8f){});
        s0     = wmma_bf16(qa1, ld_tile(kp0 + 32), s0);
        v8f s1 = wmma_bf16(qa0, ld_tile(kp1), (v8f){});
        s1     = wmma_bf16(qa1, ld_tile(kp1 + 32), s1);

        // --- scale + causal mask + row max
        float e0[8], e1[8], rmax[8];
        const int kk0 = j + lidx, kk1 = j + 16 + lidx;
#pragma unroll
        for (int r = 0; r < 8; ++r) {
            const int qi = qTile * 16 + r + 8 * half;
            e0[r] = (kk0 <= qi) ? s0[r] * SCALE : -1.0e30f;
            e1[r] = (kk1 <= qi) ? s1[r] * SCALE : -1.0e30f;
            rmax[r] = fmaxf(e0[r], e1[r]);
        }
#pragma unroll
        for (int m = 1; m < 16; m <<= 1)
#pragma unroll
            for (int r = 0; r < 8; ++r)
                rmax[r] = fmaxf(rmax[r], __shfl_xor(rmax[r], m, 16));

        // --- online softmax update; write P (bf16) to LDS in C-layout coords
        float psum[8];
#pragma unroll
        for (int r = 0; r < 8; ++r) {
            const float mnew  = fmaxf(mrun[r], rmax[r]);
            const float alpha = __expf(mrun[r] - mnew);
            mrun[r] = mnew;
#pragma unroll
            for (int t = 0; t < 4; ++t) o[t][r] *= alpha;
            const float p0 = __expf(e0[r] - mnew);
            const float p1 = __expf(e1[r] - mnew);
            psum[r] = p0 + p1;
            const int m = r + 8 * half;
            pT[m * 32 + lidx]      = (__bf16)p0;
            pT[m * 32 + 16 + lidx] = (__bf16)p1;
            lrun[r] = lrun[r] * alpha;
        }
#pragma unroll
        for (int m = 1; m < 16; m <<= 1)
#pragma unroll
            for (int r = 0; r < 8; ++r)
                psum[r] += __shfl_xor(psum[r], m, 16);
#pragma unroll
        for (int r = 0; r < 8; ++r) lrun[r] += psum[r];

        // --- stage V tile transposed into LDS: vT[dim][key]
        const __bf16* vp = Vb + ((size_t)(batch * LSEQ + j)) * EMB + head * HDIM;
#pragma unroll
        for (int c = 0; c < 8; ++c) {
            const int flat = c * 32 + lane;        // 256 chunks of 8 dims
            const int key  = flat >> 3;
            const int dc   = (flat & 7) * 8;
            v8bf vv = *(const v8bf*)(vp + (size_t)key * EMB + dc);
#pragma unroll
            for (int i = 0; i < 8; ++i) vT[(dc + i) * 32 + key] = vv[i];
        }

        // CDNA5 split dep-counter: drain LDS stores before transposed reloads
        asm volatile("s_wait_dscnt 0x0" ::: "memory");

        // --- O += P * V  (P reloaded in A-layout, V tiles in B-layout)
        const v16bf pA = ld_tile(pT + lidx * 32 + 8 * half);
#pragma unroll
        for (int t = 0; t < 4; ++t) {
            const int dim = t * 16 + lidx;
            v16bf vB = ld_tile(vT + dim * 32 + 8 * half);
            o[t] = wmma_bf16(pA, vB, o[t]);
        }
    }

    // --- normalize and write context (bf16) for the output projection
#pragma unroll
    for (int r = 0; r < 8; ++r) {
        const float inv = 1.0f / lrun[r];
        const int qrow = qTile * 16 + r + 8 * half;
#pragma unroll
        for (int t = 0; t < 4; ++t) {
            const int col = head * HDIM + t * 16 + lidx;
            ctx[((size_t)(batch * LSEQ + qrow)) * EMB + col] = (__bf16)(o[t][r] * inv);
        }
    }
}

// ---------------------------------------------------------------------------
// Host-side orchestration (graph-capture safe: kernel launches only)
// ---------------------------------------------------------------------------
extern "C" void kernel_launch(void* const* d_in, const int* in_sizes, int n_in,
                              void* d_out, int out_size, void* d_ws, size_t ws_size,
                              hipStream_t stream) {
    const float* values = (const float*)d_in[0];
    const float* keys   = (const float*)d_in[1];
    const float* query  = (const float*)d_in[2];
    // d_in[3] = mask (causal; applied analytically in attn_fwd)
    const float* Wv = (const float*)d_in[4];
    const float* bv = (const float*)d_in[5];
    const float* Wk = (const float*)d_in[6];
    const float* bk = (const float*)d_in[7];
    const float* Wq = (const float*)d_in[8];
    const float* bq = (const float*)d_in[9];
    const float* Wo = (const float*)d_in[10];
    const float* bo = (const float*)d_in[11];
    float* out = (float*)d_out;

    const size_t ACT = (size_t)ROWS * EMB;   // 4M elems
    const size_t WEL = (size_t)EMB * EMB;    // 1M elems
    __bf16* ws = (__bf16*)d_ws;
    __bf16* Xv  = ws;                 // activations, bf16
    __bf16* Xk  = Xv  + ACT;
    __bf16* Xq  = Xk  + ACT;
    __bf16* Wvb = Xq  + ACT;          // weights, bf16
    __bf16* Wkb = Wvb + WEL;
    __bf16* Wqb = Wkb + WEL;
    __bf16* Wob = Wqb + WEL;
    __bf16* Vp  = Wob + WEL;          // projected V/K/Q, bf16
    __bf16* Kp  = Vp  + ACT;
    __bf16* Qp  = Kp  + ACT;
    __bf16* Ctx = Qp  + ACT;          // attention output, bf16

    // 1) fp32 -> bf16 conversions
    const int actBlk = (int)(ACT / 4 / 256), wBlk = (int)(WEL / 4 / 256);
    cvt_f32_bf16<<<actBlk, 256, 0, stream>>>(values, Xv, (int)ACT);
    cvt_f32_bf16<<<actBlk, 256, 0, stream>>>(keys,   Xk, (int)ACT);
    cvt_f32_bf16<<<actBlk, 256, 0, stream>>>(query,  Xq, (int)ACT);
    cvt_f32_bf16<<<wBlk, 256, 0, stream>>>(Wv, Wvb, (int)WEL);
    cvt_f32_bf16<<<wBlk, 256, 0, stream>>>(Wk, Wkb, (int)WEL);
    cvt_f32_bf16<<<wBlk, 256, 0, stream>>>(Wq, Wqb, (int)WEL);
    cvt_f32_bf16<<<wBlk, 256, 0, stream>>>(Wo, Wob, (int)WEL);

    // 2) QKV projections: (4096/128 row blocks)*(1024/64 col strips) = 512 blocks
    const int gemmBlocks = (ROWS / 128) * (EMB / 64);
    gemm_bf16<<<gemmBlocks, 256, 0, stream>>>(Xv, Wvb, bv, Vp, nullptr, ROWS, EMB, EMB);
    gemm_bf16<<<gemmBlocks, 256, 0, stream>>>(Xk, Wkb, bk, Kp, nullptr, ROWS, EMB, EMB);
    gemm_bf16<<<gemmBlocks, 256, 0, stream>>>(Xq, Wqb, bq, Qp, nullptr, ROWS, EMB, EMB);

    // 3) attention: NB*NHEAD*(L/16) = 4096 waves -> 512 blocks
    attn_fwd<<<NB * NHEAD * (LSEQ / 16) * 32 / 256, 256, 0, stream>>>(Qp, Kp, Vp, Ctx);

    // 4) output projection -> fp32 d_out
    gemm_bf16<<<gemmBlocks, 256, 0, stream>>>(Ctx, Wob, bo, nullptr, out, ROWS, EMB, EMB);
}
